// GcnInfomax_86947317940936
// MI455X (gfx1250) — matrix operand order repaired and checked by reading, not compiled, not run on hardware
//
#include <hip/hip_runtime.h>
#include <math.h>

// ---------------------------------------------------------------------------
// Types for CDNA5 WMMA (wave32): v16bf A/B fragments, v8f C/D fragment
// ---------------------------------------------------------------------------
typedef __bf16 bf16_t;
typedef __attribute__((ext_vector_type(16))) __bf16 v16bf;
typedef __attribute__((ext_vector_type(8)))  float  v8f;

union ABFrag { uint4 q[2]; v16bf v; };

__device__ __forceinline__ unsigned short f2bf(float f) {
    unsigned int u = __float_as_uint(f);
    unsigned int r = (u + 0x7FFFu + ((u >> 16) & 1u)) >> 16;  // RNE
    return (unsigned short)r;
}

#define FLAG_BIAS 1
#define FLAG_RELU 2
#define FLAG_ADD  4
#define FLAG_BF16 8

#define LOG2C 0.6931471805599453f

// ---------------------------------------------------------------------------
// Edge scatter: agg[dst] += h[src]   (one wave per edge, float4 per lane)
// ---------------------------------------------------------------------------
__global__ void k_scatter(const float* __restrict__ h, int hld,
                          const int* __restrict__ esrc, const int* __restrict__ edst,
                          float* __restrict__ agg, int nE) {
    int t = blockIdx.x * blockDim.x + threadIdx.x;
    int e = t >> 5;
    int lane = t & 31;
    if (e >= nE) return;
    int s = esrc[e], d = edst[e];
    const float4 v = *(const float4*)(h + (size_t)s * hld + lane * 4);
    float* dp = agg + (size_t)d * 128 + lane * 4;
    atomicAdd(dp + 0, v.x); atomicAdd(dp + 1, v.y);
    atomicAdd(dp + 2, v.z); atomicAdd(dp + 3, v.w);
}

// zsum_bf16 = bf16(h + agg)
__global__ void k_addconv(const float* __restrict__ h, int hld,
                          const float* __restrict__ agg,
                          unsigned short* __restrict__ outb, int nRows) {
    int i = blockIdx.x * blockDim.x + threadIdx.x;
    if (i >= nRows * 128) return;
    int r = i >> 7, c = i & 127;
    outb[i] = f2bf(h[(size_t)r * hld + c] + agg[i]);
}

// f32 -> bf16 convert
__global__ void k_f2bf(const float* __restrict__ a, unsigned short* __restrict__ o, int n) {
    int i = blockIdx.x * blockDim.x + threadIdx.x;
    if (i < n) o[i] = f2bf(a[i]);
}

// ---------------------------------------------------------------------------
// Pack a (possibly transposed) f32 [K x N] weight matrix into WMMA B-fragment
// order: element (s, ntile, lane, e) <- B[k*skk + n*snn],
//   n = ntile*16 + lane%16,  k = s*32 + (e<8 ? 8*(lane/16)+e : 16+8*(lane/16)+e-8)
// so each GEMM lane reads 16 contiguous bf16 per k-step per n-tile.
// ---------------------------------------------------------------------------
__global__ void k_packb(const float* __restrict__ B, int skk, int snn,
                        int Kdim, int Ncols,
                        unsigned short* __restrict__ out, int Ksteps, int Ntiles) {
    int t = blockIdx.x * blockDim.x + threadIdx.x;
    int total = Ksteps * Ntiles * 32;
    if (t >= total) return;
    int lane = t & 31;
    int tt = (t >> 5) % Ntiles;
    int s  = (t >> 5) / Ntiles;
    int half = lane >> 4;
    int n = tt * 16 + (lane & 15);
    unsigned short* o = out + (size_t)t * 16;
#pragma unroll
    for (int e = 0; e < 16; ++e) {
        int k = s * 32 + (e < 8 ? 8 * half + e : 16 + 8 * half + (e - 8));
        float v = (k < Kdim && n < Ncols) ? B[(size_t)k * skk + (size_t)n * snn] : 0.f;
        o[e] = f2bf(v);
    }
}

// ---------------------------------------------------------------------------
// Fragment loaders (A: two contiguous 16B chunks; B: packed, 32B contiguous)
// ---------------------------------------------------------------------------
__device__ __forceinline__ void loadA(const uint4* ap0, int s, ABFrag& a) {
    const uint4* ap = ap0 + s * 4;   // s*32 ushorts = s*4 uint4
    a.q[0] = ap[0];                  // K = s*32 + 8*half + 0..7
    a.q[1] = ap[2];                  // K = s*32 + 8*half + 16..23
}
__device__ __forceinline__ void loadB4(const unsigned short* bptr, size_t bstride,
                                       int s, ABFrag* b) {
    const unsigned short* bs = bptr + s * bstride;
#pragma unroll
    for (int j = 0; j < 4; ++j) {
        const uint4* bp = (const uint4*)(bs + j * (32 * 16));
        b[j].q[0] = bp[0];
        b[j].q[1] = bp[1];
    }
}

// ---------------------------------------------------------------------------
// WMMA GEMM: C[M x N] = A[M x K](bf16,row-major) * Bpacked + epilogue
// 4 waves / block; each wave computes a 16x64 strip (4 C tiles along N),
// reusing one A fragment across 4 v_wmma_f32_16x16x32_bf16 per k-step.
// Explicit double-buffered k-loop: WMMAs consume registers loaded last
// iteration while this iteration's 10 loads are in flight.
// Requires N % 64 == 0 (all call sites: 128/384/512).
// ---------------------------------------------------------------------------
__global__ __launch_bounds__(128)
void k_gemm(const unsigned short* __restrict__ A, int lda,
            const unsigned short* __restrict__ Bp,
            const float* __restrict__ bias,
            const float* __restrict__ addsrc,
            float* __restrict__ C, unsigned short* __restrict__ Cb,
            int ldc, int Mrows, int Ksteps, int Ntiles, int flags) {
    int wave = threadIdx.x >> 5;
    int lane = threadIdx.x & 31;
    int mtile = blockIdx.x * 4 + wave;
    int ntile0 = blockIdx.y * 4;
    if (mtile * 16 >= Mrows) return;           // wave-uniform: EXEC stays full
    int half = lane >> 4;

    const unsigned short* arow = A + (size_t)(mtile * 16 + (lane & 15)) * lda + 8 * half;
    const uint4* ap0 = (const uint4*)arow;
    const size_t bstride = (size_t)Ntiles * 32 * 16;       // per k-step
    const unsigned short* bptr = Bp + ((size_t)ntile0 * 32 + lane) * 16;

    v8f acc[4] = {v8f{}, v8f{}, v8f{}, v8f{}};
    ABFrag a0, b0[4];
    loadA(ap0, 0, a0);
    loadB4(bptr, bstride, 0, b0);
    for (int s = 0; s < Ksteps; ++s) {
        ABFrag a1, b1[4];
        int sn = (s + 1 < Ksteps) ? s + 1 : s;  // last iter: dead reload, no OOB
        loadA(ap0, sn, a1);
        loadB4(bptr, bstride, sn, b1);
#pragma unroll
        for (int j = 0; j < 4; ++j)
            acc[j] = __builtin_amdgcn_wmma_f32_16x16x32_bf16(false, a0.v, false, b0[j].v,
                                                             (short)0, acc[j], false, false);
        a0 = a1;
#pragma unroll
        for (int j = 0; j < 4; ++j) b0[j] = b1[j];
    }

#pragma unroll
    for (int j = 0; j < 4; ++j) {
        int col = (ntile0 + j) * 16 + (lane & 15);
        float bv = (flags & FLAG_BIAS) ? bias[col] : 0.f;
#pragma unroll
        for (int v = 0; v < 8; ++v) {
            int row = mtile * 16 + v + 8 * half;
            if (row >= Mrows) continue;
            float c = acc[j][v] + bv;
            if (flags & FLAG_ADD)  c += addsrc[(size_t)row * ldc + col];
            if (flags & FLAG_RELU) c = fmaxf(c, 0.f);
            C[(size_t)row * ldc + col] = c;
            if (flags & FLAG_BF16) Cb[(size_t)row * ldc + col] = f2bf(c);
        }
    }
}

// ---------------------------------------------------------------------------
// WMMA GEMM fused with JSD loss epilogue: never materializes res[N x G].
//   pos (batch[i]==g): S_pos += LOG2 - softplus(-r)
//   neg:               S_neg += softplus(-r) + r - LOG2
// (masked entries contribute exactly 0 in the reference, so they are skipped)
// Same 16x64-per-wave blocking + double-buffered k-loop as k_gemm.
// ---------------------------------------------------------------------------
__global__ __launch_bounds__(128)
void k_gemm_loss(const unsigned short* __restrict__ A, int lda,
                 const unsigned short* __restrict__ Bp,
                 const int* __restrict__ batch,
                 float* __restrict__ accum,
                 int Mrows, int Ksteps, int Ntiles) {
    int wave = threadIdx.x >> 5;
    int lane = threadIdx.x & 31;
    int mtile = blockIdx.x * 4 + wave;
    int ntile0 = blockIdx.y * 4;
    if (mtile * 16 >= Mrows) return;
    int half = lane >> 4;

    const unsigned short* arow = A + (size_t)(mtile * 16 + (lane & 15)) * lda + 8 * half;
    const uint4* ap0 = (const uint4*)arow;
    const size_t bstride = (size_t)Ntiles * 32 * 16;
    const unsigned short* bptr = Bp + ((size_t)ntile0 * 32 + lane) * 16;

    v8f acc[4] = {v8f{}, v8f{}, v8f{}, v8f{}};
    ABFrag a0, b0[4];
    loadA(ap0, 0, a0);
    loadB4(bptr, bstride, 0, b0);
    for (int s = 0; s < Ksteps; ++s) {
        ABFrag a1, b1[4];
        int sn = (s + 1 < Ksteps) ? s + 1 : s;
        loadA(ap0, sn, a1);
        loadB4(bptr, bstride, sn, b1);
#pragma unroll
        for (int j = 0; j < 4; ++j)
            acc[j] = __builtin_amdgcn_wmma_f32_16x16x32_bf16(false, a0.v, false, b0[j].v,
                                                             (short)0, acc[j], false, false);
        a0 = a1;
#pragma unroll
        for (int j = 0; j < 4; ++j) b0[j] = b1[j];
    }

    float s_pos = 0.f, s_neg = 0.f;
#pragma unroll
    for (int j = 0; j < 4; ++j) {
        int col = (ntile0 + j) * 16 + (lane & 15);
#pragma unroll
        for (int v = 0; v < 8; ++v) {
            int row = mtile * 16 + v + 8 * half;
            if (row >= Mrows) continue;
            float r = acc[j][v];
            // softplus(-r) = max(-r,0) + log1p(exp(-|r|))
            float sp = fmaxf(-r, 0.f) + log1pf(expf(-fabsf(r)));
            if (batch[row] == col) s_pos += LOG2C - sp;
            else                   s_neg += sp + r - LOG2C;
        }
    }
#pragma unroll
    for (int o = 16; o > 0; o >>= 1) {
        s_pos += __shfl_down(s_pos, o);
        s_neg += __shfl_down(s_neg, o);
    }
    if (lane == 0) {
        atomicAdd(&accum[0], s_pos);
        atomicAdd(&accum[1], s_neg);
    }
}

// ---------------------------------------------------------------------------
// BatchNorm (training mode, biased stats): one block per channel
// ---------------------------------------------------------------------------
__global__ __launch_bounds__(256)
void k_bnstats(const float* __restrict__ z, const float* __restrict__ gamma,
               float* __restrict__ mu, float* __restrict__ rs, int nRows) {
    __shared__ float s1[256], s2[256];
    int c = blockIdx.x;  // 0..127
    float a = 0.f, b = 0.f;
    for (int r = threadIdx.x; r < nRows; r += 256) {
        float v = z[(size_t)r * 128 + c];
        a += v; b += v * v;
    }
    s1[threadIdx.x] = a; s2[threadIdx.x] = b;
    __syncthreads();
    for (int o = 128; o > 0; o >>= 1) {
        if ((int)threadIdx.x < o) {
            s1[threadIdx.x] += s1[threadIdx.x + o];
            s2[threadIdx.x] += s2[threadIdx.x + o];
        }
        __syncthreads();
    }
    if (threadIdx.x == 0) {
        float m = s1[0] / (float)nRows;
        float var = s2[0] / (float)nRows - m * m;
        mu[c] = m;
        rs[c] = gamma[c] * rsqrtf(var + 1e-5f);
    }
}

// normalize + write f32/bf16 slice of M + pool scatter (all fused)
__global__ void k_bnapply(const float* __restrict__ z,
                          const float* __restrict__ mu, const float* __restrict__ rs,
                          const float* __restrict__ beta,
                          const int* __restrict__ batch,
                          float* __restrict__ Mout, unsigned short* __restrict__ Mbout,
                          float* __restrict__ pools, int nRows) {
    int i = blockIdx.x * blockDim.x + threadIdx.x;
    if (i >= nRows * 128) return;
    int r = i >> 7, c = i & 127;
    float v = (z[i] - mu[c]) * rs[c] + beta[c];
    Mout[(size_t)r * 384 + c] = v;
    Mbout[(size_t)r * 384 + c] = f2bf(v);
    atomicAdd(&pools[(size_t)batch[r] * 384 + c], v);
}

__global__ void k_final(const float* __restrict__ acc, float* __restrict__ out,
                        float inv_n, float inv_neg) {
    out[0] = acc[1] * inv_neg - acc[0] * inv_n;   // E_neg - E_pos
}

// ---------------------------------------------------------------------------
// Host orchestration
// ---------------------------------------------------------------------------
extern "C" void kernel_launch(void* const* d_in, const int* in_sizes, int n_in,
                              void* d_out, int out_size, void* d_ws, size_t ws_size,
                              hipStream_t stream) {
    const int H = 128, EMB = 384, G = 512, NLAY = 3;
    const int N = in_sizes[0] / H;       // 50000
    const int E = in_sizes[1] / 2;       // 800000

    const float* x        = (const float*)d_in[0];
    const int*   ei       = (const int*)d_in[1];
    const int*   batch    = (const int*)d_in[2];
    const float* gin_w1   = (const float*)d_in[4];
    const float* gin_b1   = (const float*)d_in[5];
    const float* gin_w2   = (const float*)d_in[6];
    const float* gin_b2   = (const float*)d_in[7];
    const float* bn_gamma = (const float*)d_in[8];
    const float* bn_beta  = (const float*)d_in[9];
    const float* lw[4] = { (const float*)d_in[10], (const float*)d_in[12],
                           (const float*)d_in[14], (const float*)d_in[16] };
    const float* lb[4] = { (const float*)d_in[11], (const float*)d_in[13],
                           (const float*)d_in[15], (const float*)d_in[17] };
    const float* gw[4] = { (const float*)d_in[18], (const float*)d_in[20],
                           (const float*)d_in[22], (const float*)d_in[24] };
    const float* gb[4] = { (const float*)d_in[19], (const float*)d_in[21],
                           (const float*)d_in[23], (const float*)d_in[25] };

    char* ws = (char*)d_ws;
    const size_t szNH  = (size_t)N * H * 4;
    const size_t szNHb = (size_t)N * H * 2;
    const size_t szNE  = (size_t)N * EMB * 4;
    const size_t szNEb = (size_t)N * EMB * 2;
    const size_t szGE  = (size_t)G * EMB * 4;
    const size_t szGEb = (size_t)G * EMB * 2;

    // Region A: GIN scratch (102.4MB), later reused as FF ping buffer (115.2MB)
    const size_t oAgg   = 0;
    const size_t oZsumb = oAgg + szNH;
    const size_t oT1    = oZsumb + szNHb;
    const size_t oT1b   = oT1 + szNH;
    const size_t oZ     = oT1b + szNHb;
    const size_t oBufP  = 0;
    const size_t oBufPb = szNE;
    const size_t oEndA  = szNE + szNEb;          // 115.2MB  (>= GIN layout)
    const size_t oM     = oEndA;
    const size_t oMb    = oM + szNE;
    const size_t oBufQ  = oMb + szNEb;
    const size_t oBufQb = oBufQ + szNE;
    const size_t oPools = oBufQb + szNEb;
    const size_t oPoolsb= oPools + szGE;
    const size_t oGP    = oPoolsb + szGEb;
    const size_t oGPb   = oGP + szGE;
    const size_t oGQ    = oGPb + szGEb;
    const size_t oGQb   = oGQ + szGE;
    const size_t oMu    = oGQb + szGEb;
    const size_t oRs    = oMu + 512;
    const size_t oAcc   = oRs + 512;
    const size_t oPackB = oAcc + 256;            // max: 12*32*32*16*2 = 384KB
    (void)ws_size; (void)n_in; (void)out_size;

    float*          agg    = (float*)(ws + oAgg);
    unsigned short* zsumb  = (unsigned short*)(ws + oZsumb);
    float*          t1     = (float*)(ws + oT1);
    unsigned short* t1b    = (unsigned short*)(ws + oT1b);
    float*          zbuf   = (float*)(ws + oZ);
    float*          Mf     = (float*)(ws + oM);
    unsigned short* Mb16   = (unsigned short*)(ws + oMb);
    float*          bufP   = (float*)(ws + oBufP);
    unsigned short* bufPb  = (unsigned short*)(ws + oBufPb);
    float*          bufQ   = (float*)(ws + oBufQ);
    unsigned short* bufQb  = (unsigned short*)(ws + oBufQb);
    float*          pools  = (float*)(ws + oPools);
    unsigned short* poolsb = (unsigned short*)(ws + oPoolsb);
    float*          gP     = (float*)(ws + oGP);
    unsigned short* gPb    = (unsigned short*)(ws + oGPb);
    float*          gQ     = (float*)(ws + oGQ);
    unsigned short* gQb    = (unsigned short*)(ws + oGQb);
    float*          mu     = (float*)(ws + oMu);
    float*          rs     = (float*)(ws + oRs);
    float*          accF   = (float*)(ws + oAcc);
    unsigned short* packB  = (unsigned short*)(ws + oPackB);

    auto pack = [&](const float* B, int skk, int snn, int K, int Nc) {
        int Ks = (K + 31) / 32, Nt = (Nc + 15) / 16;
        int tot = Ks * Nt * 32;
        k_packb<<<dim3((tot + 255) / 256), dim3(256), 0, stream>>>(
            B, skk, snn, K, Nc, packB, Ks, Nt);
    };
    auto gemm = [&](const unsigned short* A, int lda, const float* bias,
                    const float* addsrc, float* C, unsigned short* Cb,
                    int ldc, int Mr, int K, int Nc, int flags) {
        int Ks = K / 32, Nt = Nc / 16;            // Nc % 64 == 0 at all call sites
        int mt = (Mr + 15) / 16;
        k_gemm<<<dim3((mt + 3) / 4, Nt / 4), dim3(128), 0, stream>>>(
            A, lda, packB, bias, addsrc, C, Cb, ldc, Mr, Ks, Nt, flags);
    };

    hipMemsetAsync(ws + oPools, 0, szGE, stream);

    // ---------------- GIN layers ----------------
    for (int l = 0; l < NLAY; ++l) {
        const float* hptr = (l == 0) ? x : (Mf + (size_t)(l - 1) * H);
        int hld = (l == 0) ? H : EMB;

        hipMemsetAsync(ws + oAgg, 0, szNH, stream);
        k_scatter<<<dim3((E * 32 + 255) / 256), dim3(256), 0, stream>>>(
            hptr, hld, ei, ei + E, agg, E);
        k_addconv<<<dim3((N * H + 255) / 256), dim3(256), 0, stream>>>(
            hptr, hld, agg, zsumb, N);

        pack(gin_w1 + (size_t)l * H * H, H, 1, H, H);
        gemm(zsumb, H, gin_b1 + l * H, nullptr, t1, t1b, H, N, H, H,
             FLAG_BIAS | FLAG_RELU | FLAG_BF16);
        pack(gin_w2 + (size_t)l * H * H, H, 1, H, H);
        gemm(t1b, H, gin_b2 + l * H, nullptr, zbuf, nullptr, H, N, H, H,
             FLAG_BIAS | FLAG_RELU);

        k_bnstats<<<dim3(H), dim3(256), 0, stream>>>(zbuf, bn_gamma + l * H, mu, rs, N);
        k_bnapply<<<dim3((N * H + 255) / 256), dim3(256), 0, stream>>>(
            zbuf, mu, rs, bn_beta + l * H, batch,
            Mf + (size_t)l * H, Mb16 + (size_t)l * H, pools + (size_t)l * H, N);
    }

    // ---------------- g_enc = FF(pools) ----------------
    k_f2bf<<<dim3((G * EMB + 255) / 256), dim3(256), 0, stream>>>(pools, poolsb, G * EMB);
    pack(gw[0], EMB, 1, EMB, EMB);
    gemm(poolsb, EMB, gb[0], nullptr, gP, gPb, EMB, G, EMB, EMB,
         FLAG_BIAS | FLAG_RELU | FLAG_BF16);
    pack(gw[1], EMB, 1, EMB, EMB);
    gemm(gPb, EMB, gb[1], nullptr, gQ, gQb, EMB, G, EMB, EMB,
         FLAG_BIAS | FLAG_RELU | FLAG_BF16);
    pack(gw[2], EMB, 1, EMB, EMB);
    gemm(gQb, EMB, gb[2], nullptr, gP, nullptr, EMB, G, EMB, EMB,
         FLAG_BIAS | FLAG_RELU);
    pack(gw[3], EMB, 1, EMB, EMB);
    gemm(poolsb, EMB, gb[3], gP, gQ /* = g_enc f32 */, nullptr, EMB, G, EMB, EMB,
         FLAG_BIAS | FLAG_ADD);

    // ---------------- l_enc = FF(M) ----------------
    pack(lw[0], EMB, 1, EMB, EMB);
    gemm(Mb16, EMB, lb[0], nullptr, bufP, bufPb, EMB, N, EMB, EMB,
         FLAG_BIAS | FLAG_RELU | FLAG_BF16);
    pack(lw[1], EMB, 1, EMB, EMB);
    gemm(bufPb, EMB, lb[1], nullptr, bufQ, bufQb, EMB, N, EMB, EMB,
         FLAG_BIAS | FLAG_RELU | FLAG_BF16);
    pack(lw[2], EMB, 1, EMB, EMB);
    gemm(bufQb, EMB, lb[2], nullptr, bufP, nullptr, EMB, N, EMB, EMB,
         FLAG_BIAS | FLAG_RELU);
    pack(lw[3], EMB, 1, EMB, EMB);
    gemm(Mb16, EMB, lb[3], bufP, bufQ /* = l_enc f32 */, bufQb /* l_enc bf16 */,
         EMB, N, EMB, EMB, FLAG_BIAS | FLAG_ADD | FLAG_BF16);

    // ---------------- fused res = l_enc @ g_enc^T + JSD loss ----------------
    // B(k,n) = g_enc[n*EMB + k]  -> skk=1, snn=EMB
    pack(gQ, 1, EMB, EMB, G);
    hipMemsetAsync(ws + oAcc, 0, 2 * sizeof(float), stream);
    {
        int mt = (N + 15) / 16;
        k_gemm_loss<<<dim3((mt + 3) / 4, (G / 16) / 4), dim3(128), 0, stream>>>(
            bufQb, EMB, packB, batch, accF, N, EMB / 32, G / 16);
    }
    k_final<<<dim3(1), dim3(1), 0, stream>>>(
        accF, (float*)d_out, 1.f / (float)N,
        (float)(1.0 / ((double)N * (double)(G - 1))));
}
